// ResNet_26173530702253
// MI455X (gfx1250) — matrix operand order repaired and checked by reading, not compiled, not run on hardware
//
#include <hip/hip_runtime.h>
#include <hip/hip_bf16.h>

typedef __attribute__((ext_vector_type(16))) _Float16 v16h;
typedef __attribute__((ext_vector_type(8)))  _Float16 v8h;
typedef __attribute__((ext_vector_type(4)))  _Float16 v4h;
typedef __attribute__((ext_vector_type(8)))  float    v8f;

#define N_NODES 50000
#define N_EDGES 800000
#define HID     64
#define BLOCK   128
#define WAVES   4
#define NTILES  (N_EDGES / 16)   // 50000 full 16-edge tiles
#define GRID_EDGE 1250           // 1250*4 = 5000 waves -> 10 tiles each, no remainder

// ---------------- helpers ----------------

// monotone float -> u32 key so unsigned atomicMax == float max (total order)
// branchless: neg -> ~u, non-neg -> u | 0x80000000
__device__ __forceinline__ unsigned fkey(float f) {
  unsigned u = __float_as_uint(f);
  return u ^ ((unsigned)((int)u >> 31) | 0x80000000u);
}

__device__ __forceinline__ v8f wmma16(v16h a, v16h b, v8f c) {
  // emits v_wmma_f32_16x16x32_f16
  return __builtin_amdgcn_wmma_f32_16x16x32_f16(false, a, false, b, (short)0, c,
                                                false, false);
}

// A fragment (16-bit A 16x32, ISA 7.12.2): lane<16 holds K {k0..k0+7, k0+16..k0+23},
// lane>=16 holds K {k0+8..k0+15, k0+24..k0+31}. Caller passes koff = ktile*32 + hi*8.
__device__ __forceinline__ v16h load_afrag(const _Float16* ms, int row, int koff, int ldk) {
  const _Float16* p = ms + row * ldk + koff;
  v8h c0 = *(const v8h*)p;         // 16B ds_load
  v8h c1 = *(const v8h*)(p + 16);  // 16B ds_load
  v16h a;
#pragma unroll
  for (int i = 0; i < 8; ++i) { a[i] = c0[i]; a[i + 8] = c1[i]; }
  return a;
}

// B fragment: pre-swizzled in LDS, 16 contiguous halfs per lane
__device__ __forceinline__ v16h load_bfrag(const _Float16* p) {
  v8h c0 = *(const v8h*)p;
  v8h c1 = *(const v8h*)(p + 8);
  v16h b;
#pragma unroll
  for (int i = 0; i < 8; ++i) { b[i] = c0[i]; b[i + 8] = c1[i]; }
  return b;
}

// LayerNorm affine + LeakyReLU(0.2); leaky(t) == max(t, 0.2t) since slope < 1
__device__ __forceinline__ float lnact(float x, float mu, float rs, float g, float b) {
  float t = (x - mu) * rs * g + b;
  return fmaxf(t, 0.2f * t);
}

// Stage weight matrix w[64][K] (torch [out,in]) into LDS B-fragment layout.
// Fragment (kt,nt): lane l, half h -> B[k][n] = w[n][k],
// n = nt*16 + (l&15), k = kt*32 + ((l&16)>>1) + (h&7) + ((h&8)<<1)   (dense 32x16 B layout, ISA 7.12.4)
__device__ __forceinline__ void stage_w(_Float16* wf, const float* __restrict__ w,
                                        int K, int KT, int tid) {
  int total = KT * 4 * 32 * 16;
  for (int i = tid; i < total; i += BLOCK) {
    int h = i & 15;
    int l = (i >> 4) & 31;
    int f = i >> 9;
    int kt = f >> 2, nt = f & 3;
    int n = nt * 16 + (l & 15);
    int k = kt * 32 + ((l & 16) >> 1) + (h & 7) + ((h & 8) << 1);
    wf[i] = (_Float16)w[n * K + k];
  }
}

// LayerNorm(64) + LeakyReLU on a C fragment set (4 n-tiles), restage as f16 A input.
// C layout: acc[nt][r] = out[M = r + hi*8][n = nt*16 + lo]
__device__ __forceinline__ void ln_restage(const v8f* acc, _Float16* ms,
                                           const float* sg, const float* sb, int lane) {
  int hi = lane >> 4, lo = lane & 15;
  float mu[8], rs[8];
#pragma unroll
  for (int r = 0; r < 8; ++r) {
    float s = acc[0][r] + acc[1][r] + acc[2][r] + acc[3][r];
    float q = acc[0][r] * acc[0][r] + acc[1][r] * acc[1][r] +
              acc[2][r] * acc[2][r] + acc[3][r] * acc[3][r];
#pragma unroll
    for (int m = 1; m < 16; m <<= 1) {  // butterfly within 16-lane half
      s += __shfl_xor(s, m, 32);
      q += __shfl_xor(q, m, 32);
    }
    float mean = s * (1.f / 64.f);
    float var  = q * (1.f / 64.f) - mean * mean;
    mu[r] = mean;
    rs[r] = rsqrtf(var + 1e-5f);
  }
#pragma unroll
  for (int nt = 0; nt < 4; ++nt) {
    int n = nt * 16 + lo;
    float g = sg[n], b = sb[n];
#pragma unroll
    for (int r = 0; r < 8; ++r) {
      float t = lnact(acc[nt][r], mu[r], rs[r], g, b);
      ms[(hi * 8 + r) * 64 + n] = (_Float16)t;
    }
  }
}

// ---------------- kernels ----------------

__global__ void init_keys_kernel(unsigned* __restrict__ keys) {
  int i = blockIdx.x * blockDim.x + threadIdx.x;
  if (i < N_NODES * HID) keys[i] = 0x007FFFFFu;  // fkey(-inf)
}

__global__ void __launch_bounds__(BLOCK)
edge_mlp_wmma_kernel(const float* __restrict__ x, const float* __restrict__ ew,
                     const long long* __restrict__ ei,
                     const float* __restrict__ w1, const float* __restrict__ w2,
                     const float* __restrict__ w3,
                     const float* __restrict__ g1, const float* __restrict__ b1,
                     const float* __restrict__ g2, const float* __restrict__ b2,
                     const float* __restrict__ g3, const float* __restrict__ b3,
                     unsigned* __restrict__ outkeys) {
  __shared__ _Float16 wf1[16 * 32 * 16];  // layer1: 4 ktiles x 4 ntiles   (16 KB)
  __shared__ _Float16 wf2[8 * 32 * 16];   // layer2: 2 x 4                 (8 KB)
  __shared__ _Float16 wf3[8 * 32 * 16];   // layer3: 2 x 4                 (8 KB)
  __shared__ float sg1[128], sb1[128], sg2[64], sb2[64], sg3[64], sb3[64];
  __shared__ _Float16 mstage[WAVES][16 * 128];  // per-wave staging (16 KB)

  const int tid = threadIdx.x;
  stage_w(wf1, w1, 128, 4, tid);
  stage_w(wf2, w2, 64, 2, tid);
  stage_w(wf3, w3, 64, 2, tid);
  for (int i = tid; i < 128; i += BLOCK) { sg1[i] = g1[i]; sb1[i] = b1[i]; }
  for (int i = tid; i < 64; i += BLOCK) {
    sg2[i] = g2[i]; sb2[i] = b2[i]; sg3[i] = g3[i]; sb3[i] = b3[i];
  }
  __syncthreads();

  const int lane = tid & 31;
  const int wave = tid >> 5;
  const int hi   = lane >> 4;   // half-wave id
  const int lo   = lane & 15;
  _Float16* ms = mstage[wave];

  const int nwaves = gridDim.x * WAVES;
  const int wgid   = blockIdx.x * WAVES + wave;

  for (int tile = wgid; tile < NTILES; tile += nwaves) {
    const int e = tile * 16 + lo;  // lane pair (lo, lo+16) shares edge e
    const long long srcN = ei[e];
    const long long tgtN = ei[N_EDGES + e];
    const float wgt = ew[e];

    // ---- build edge feature: lane hi=0 -> x_i (feat 0..63); hi=1 -> w*(x_j-x_i)
    // r = c1*a + scl*b : hi=0 -> 1*a + 0*b = x_i ; hi=1 -> -w*a + w*b = w*(x_j-x_i)
    const float4* pa = (const float4*)(x + (size_t)tgtN * HID);
    const float4* pb = hi ? (const float4*)(x + (size_t)srcN * HID) : pa;
    const float c1  = hi ? -wgt : 1.f;
    const float scl = hi ? wgt : 0.f;

    float4 v[16];
    float s = 0.f, s2 = 0.f;
#pragma unroll
    for (int c = 0; c < 16; ++c) {
      float4 a = pa[c], b = pb[c];
      float4 r;
      r.x = c1 * a.x + scl * b.x;
      r.y = c1 * a.y + scl * b.y;
      r.z = c1 * a.z + scl * b.z;
      r.w = c1 * a.w + scl * b.w;
      v[c] = r;
      s  += r.x + r.y + r.z + r.w;
      s2 += r.x * r.x + r.y * r.y + r.z * r.z + r.w * r.w;
    }
    // LN over 128 dims: combine the two half-rows of this edge
    s  += __shfl_xor(s, 16, 32);
    s2 += __shfl_xor(s2, 16, 32);
    const float mu = s * (1.f / 128.f);
    const float var = s2 * (1.f / 128.f) - mu * mu;
    const float rs = rsqrtf(var + 1e-5f);

    // LN1 + LeakyReLU -> f16 row-major staging [16][128]
#pragma unroll
    for (int c = 0; c < 16; ++c) {
      int idx = hi * 64 + c * 4;
      float4 r = v[c];
      v4h h;
      h[0] = (_Float16)lnact(r.x, mu, rs, sg1[idx + 0], sb1[idx + 0]);
      h[1] = (_Float16)lnact(r.y, mu, rs, sg1[idx + 1], sb1[idx + 1]);
      h[2] = (_Float16)lnact(r.z, mu, rs, sg1[idx + 2], sb1[idx + 2]);
      h[3] = (_Float16)lnact(r.w, mu, rs, sg1[idx + 3], sb1[idx + 3]);
      *(v4h*)(ms + lo * 128 + idx) = h;
    }
    asm volatile("" ::: "memory");  // keep cross-lane LDS store->load ordered

    // ---- layer 1 GEMM: [16x128] x [128x64], 16 WMMAs
    v8f zero8 = {0.f, 0.f, 0.f, 0.f, 0.f, 0.f, 0.f, 0.f};
    v8f acc1[4] = {zero8, zero8, zero8, zero8};
#pragma unroll
    for (int kt = 0; kt < 4; ++kt) {
      v16h a = load_afrag(ms, lo, kt * 32 + hi * 8, 128);
#pragma unroll
      for (int nt = 0; nt < 4; ++nt) {
        v16h b = load_bfrag(wf1 + ((kt * 4 + nt) * 32 + lane) * 16);
        acc1[nt] = wmma16(a, b, acc1[nt]);
      }
    }

    // ---- LN2 + act, restage, layer 2 GEMM (8 WMMAs)
    ln_restage(acc1, ms, sg2, sb2, lane);
    asm volatile("" ::: "memory");
    v8f acc2[4] = {zero8, zero8, zero8, zero8};
#pragma unroll
    for (int kt = 0; kt < 2; ++kt) {
      v16h a = load_afrag(ms, lo, kt * 32 + hi * 8, 64);
#pragma unroll
      for (int nt = 0; nt < 4; ++nt) {
        v16h b = load_bfrag(wf2 + ((kt * 4 + nt) * 32 + lane) * 16);
        acc2[nt] = wmma16(a, b, acc2[nt]);
      }
    }

    // ---- LN3 + act, restage, layer 3 GEMM (8 WMMAs)
    ln_restage(acc2, ms, sg3, sb3, lane);
    asm volatile("" ::: "memory");
    v8f acc3[4] = {zero8, zero8, zero8, zero8};
#pragma unroll
    for (int kt = 0; kt < 2; ++kt) {
      v16h a = load_afrag(ms, lo, kt * 32 + hi * 8, 64);
#pragma unroll
      for (int nt = 0; nt < 4; ++nt) {
        v16h b = load_bfrag(wf3 + ((kt * 4 + nt) * 32 + lane) * 16);
        acc3[nt] = wmma16(a, b, acc3[nt]);
      }
    }

    // ---- segment_max scatter: acc3[nt][r] = out[M=r+hi*8][n=nt*16+lo]
    const int tgt32 = (int)tgtN;
#pragma unroll
    for (int r = 0; r < 8; ++r) {
      int M = hi * 8 + r;
      int t = __shfl(tgt32, M, 32);  // tgt of edge tile*16+M lives in lane M
      unsigned* rowp = outkeys + (size_t)t * HID;
#pragma unroll
      for (int nt = 0; nt < 4; ++nt) {
        atomicMax(rowp + nt * 16 + lo, fkey(acc3[nt][r]));
      }
    }
  }
}

__global__ void finalize_kernel(const float* __restrict__ x, float* __restrict__ out) {
  int i = blockIdx.x * blockDim.x + threadIdx.x;
  if (i >= N_NODES * HID) return;
  unsigned k = __float_as_uint(out[i]);  // raw key bits
  unsigned u = (k & 0x80000000u) ? (k & 0x7FFFFFFFu) : ~k;
  float f = __uint_as_float(u);
  bool finite = ((u >> 23) & 0xFFu) != 0xFFu;  // isolated nodes stay -inf -> 0
  out[i] = x[i] + (finite ? f : 0.f);
}

// ---------------- launcher ----------------

extern "C" void kernel_launch(void* const* d_in, const int* in_sizes, int n_in,
                              void* d_out, int out_size, void* d_ws, size_t ws_size,
                              hipStream_t stream) {
  const float*     x  = (const float*)d_in[0];
  const float*     ew = (const float*)d_in[1];
  const long long* ei = (const long long*)d_in[2];
  const float*     w1 = (const float*)d_in[3];
  const float*     w2 = (const float*)d_in[4];
  const float*     w3 = (const float*)d_in[5];
  const float*     g1 = (const float*)d_in[6];
  const float*     b1 = (const float*)d_in[7];
  const float*     g2 = (const float*)d_in[8];
  const float*     b2 = (const float*)d_in[9];
  const float*     g3 = (const float*)d_in[10];
  const float*     b3 = (const float*)d_in[11];

  // d_out doubles as the u32 max-key buffer (same size as output), then is
  // decoded in place by finalize -> no workspace needed, fully deterministic.
  unsigned* keys = (unsigned*)d_out;

  const int total = N_NODES * HID;
  init_keys_kernel<<<(total + 255) / 256, 256, 0, stream>>>(keys);
  edge_mlp_wmma_kernel<<<GRID_EDGE, BLOCK, 0, stream>>>(
      x, ew, ei, w1, w2, w3, g1, b1, g2, b2, g3, b3, keys);
  finalize_kernel<<<(total + 255) / 256, 256, 0, stream>>>(x, (float*)d_out);
}
